// SparseTargetAttention_85332410237228
// MI455X (gfx1250) — matrix operand order repaired and checked by reading, not compiled, not run on hardware
//
#include <hip/hip_runtime.h>
#include <hip/hip_bf16.h>

#define B_SZ   128
#define L_SZ   4096
#define D_SZ   128
#define E_SZ   64
#define NITER  50
#define NCHUNK (L_SZ / 256)   // 16 chunks per batch row in k_wsum

typedef __attribute__((ext_vector_type(16))) __bf16 v16bf;
typedef __attribute__((ext_vector_type(8)))  __bf16 v8bf;
typedef __attribute__((ext_vector_type(8)))  float  v8f;

__device__ __forceinline__ float warp_sum(float v) {
  #pragma unroll
  for (int o = 16; o > 0; o >>= 1) v += __shfl_xor(v, o, 32);
  return v;
}

// ---------------------------------------------------------------------------
// Kernel 0: c[b][d] = bias_r[d] + sum_e q[b][e]*Wr2[d][e],
//           q[b][e] = sum_d target[b][d]*Wf_w[e][d] + Wf_b[e]
// ---------------------------------------------------------------------------
__global__ __launch_bounds__(128) void k_prep(
    const float* __restrict__ target, const float* __restrict__ Wr2,
    const float* __restrict__ bias_r, const float* __restrict__ Wf_w,
    const float* __restrict__ Wf_b, float* __restrict__ c_buf)
{
  __shared__ float qs[E_SZ];
  const int b = blockIdx.x, tid = threadIdx.x;
  const float* tb = target + (size_t)b * D_SZ;
  if (tid < E_SZ) {
    float q = Wf_b[tid];
    const float* wrow = Wf_w + (size_t)tid * D_SZ;
    for (int d = 0; d < D_SZ; ++d) q += tb[d] * wrow[d];
    qs[tid] = q;
  }
  __syncthreads();
  float cv = bias_r[tid];
  const float* w2row = Wr2 + (size_t)tid * E_SZ;
  for (int e = 0; e < E_SZ; ++e) cv += qs[e] * w2row[e];
  c_buf[b * D_SZ + tid] = cv;
}

// ---------------------------------------------------------------------------
// Kernel 1: scores[b,l] = wr3 . relu(R[b,l,:] @ Wr1^T + c[b,:])
// bf16 WMMA 16x16x32, fp32 accumulate. 8 waves/WG, each wave = 16 rows.
// Wr1 staged in LDS as bf16, padded stride 136 (=> +4 bank shift per row,
// conflict-free 16B fragment loads across the 16 lanes of a half-wave).
// ---------------------------------------------------------------------------
__global__ __launch_bounds__(256) void k_scores(
    const float* __restrict__ R, const float* __restrict__ Wr1,
    const float* __restrict__ Wr3, const float* __restrict__ c_buf,
    float* __restrict__ scores)
{
  __shared__ __bf16 sW[D_SZ][136];
  const int b    = blockIdx.y;
  const int l0   = blockIdx.x * 128;
  const int tid  = threadIdx.x;
  const int wave = tid >> 5;
  const int lane = tid & 31;

  // Stage Wr1 (fp32, row-major [e][d]) -> LDS bf16
  #pragma unroll
  for (int i = 0; i < 16; ++i) {
    const int f   = tid + 256 * i;        // float4 index into 128x128
    const int row = f >> 5;
    const int c4  = (f & 31) << 2;
    const float4 v = ((const float4*)Wr1)[f];
    sW[row][c4 + 0] = (__bf16)v.x;
    sW[row][c4 + 1] = (__bf16)v.y;
    sW[row][c4 + 2] = (__bf16)v.z;
    sW[row][c4 + 3] = (__bf16)v.w;
  }
  __syncthreads();

  const int m0 = wave * 16;
  const int rl = lane & 15;       // row (A) / column (B) within tile
  const int kh = lane >> 4;       // K-half selector per WMMA layout
  const float* rowp = R + ((size_t)b * L_SZ + (size_t)(l0 + m0 + rl)) * D_SZ;

  v8f acc[8];
  #pragma unroll
  for (int nt = 0; nt < 8; ++nt)
    #pragma unroll
    for (int j = 0; j < 8; ++j) acc[nt][j] = 0.0f;

  #pragma unroll
  for (int kt = 0; kt < 4; ++kt) {
    const int kb = kt * 32;
    // A fragment, 16-bit A 16x32 layout:
    // lane<16: elems 0..7 -> K=kb+0..7,  elems 8..15 -> K=kb+16..23
    // lane>=16: elems 0..7 -> K=kb+8..15, elems 8..15 -> K=kb+24..31
    const float4 f0 = *(const float4*)(rowp + kb + kh * 8);
    const float4 f1 = *(const float4*)(rowp + kb + kh * 8 + 4);
    const float4 f2 = *(const float4*)(rowp + kb + 16 + kh * 8);
    const float4 f3 = *(const float4*)(rowp + kb + 16 + kh * 8 + 4);
    v16bf a;
    a[0]  = (__bf16)f0.x; a[1]  = (__bf16)f0.y; a[2]  = (__bf16)f0.z; a[3]  = (__bf16)f0.w;
    a[4]  = (__bf16)f1.x; a[5]  = (__bf16)f1.y; a[6]  = (__bf16)f1.z; a[7]  = (__bf16)f1.w;
    a[8]  = (__bf16)f2.x; a[9]  = (__bf16)f2.y; a[10] = (__bf16)f2.z; a[11] = (__bf16)f2.w;
    a[12] = (__bf16)f3.x; a[13] = (__bf16)f3.y; a[14] = (__bf16)f3.z; a[15] = (__bf16)f3.w;

    #pragma unroll
    for (int nt = 0; nt < 8; ++nt) {
      // B fragment (32x16): B[k][n] = Wr1[nbase+n][kb+k].
      // lane<16 holds K=kb+0..7,16..23; lane>=16 holds K=kb+8..15,24..31 (2 bf16/VGPR).
      const __bf16* wrow = &sW[nt * 16 + rl][kb + kh * 8];
      const v8bf lo = *(const v8bf*)(wrow);
      const v8bf hi = *(const v8bf*)(wrow + 16);
      const v16bf bfrag = __builtin_shufflevector(
          lo, hi, 0, 1, 2, 3, 4, 5, 6, 7, 8, 9, 10, 11, 12, 13, 14, 15);
      acc[nt] = __builtin_amdgcn_wmma_f32_16x16x32_bf16(
          false, a, false, bfrag, (short)0, acc[nt], false, false);
    }
  }

  // Epilogue: relu(+c), dot with wr3, reduce the 16 N-lanes per row.
  // C/D layout: lane<16 -> VGPR j holds M=j; lane>=16 -> M=j+8; N = lane&15.
  float sacc[8];
  #pragma unroll
  for (int j = 0; j < 8; ++j) sacc[j] = 0.0f;
  #pragma unroll
  for (int nt = 0; nt < 8; ++nt) {
    const int n   = nt * 16 + rl;
    const float wn = Wr3[n];
    const float cn = c_buf[b * D_SZ + n];
    #pragma unroll
    for (int j = 0; j < 8; ++j) {
      float h = acc[nt][j] + cn;
      h = h > 0.0f ? h : 0.0f;
      sacc[j] += wn * h;
    }
  }
  #pragma unroll
  for (int off = 1; off < 16; off <<= 1)
    #pragma unroll
    for (int j = 0; j < 8; ++j) sacc[j] += __shfl_xor(sacc[j], off, 32);

  if ((lane & 15) == 0) {
    const int base = b * L_SZ + l0 + m0 + kh * 8;
    #pragma unroll
    for (int j = 0; j < 8; ++j) scores[base + j] = sacc[j];
  }
}

// ---------------------------------------------------------------------------
// Kernel 2: alpha-entmax via 50-iteration bisection, one WG per batch row.
// Replicates the reference exactly (f_lo fixed, final p at last tau_m).
// Writes a[b,:] in place over scores[b,:].
// ---------------------------------------------------------------------------
__device__ __forceinline__ float blk_sum(float v, float* red, float* bc, int tid) {
  v = warp_sum(v);
  __syncthreads();                       // protect red/bc reuse across calls
  if ((tid & 31) == 0) red[tid >> 5] = v;
  __syncthreads();
  if (tid < 32) {
    float x = (tid < 8) ? red[tid] : 0.0f;
    x = warp_sum(x);
    if (tid == 0) *bc = x;
  }
  __syncthreads();
  return *bc;
}

__global__ __launch_bounds__(256) void k_entmax(
    const float* __restrict__ alpha, float* __restrict__ sc)
{
  __shared__ float sXs[L_SZ];
  __shared__ float red[8];
  __shared__ float bc;
  const int b = blockIdx.x, tid = threadIdx.x;
  const float am1 = alpha[b] - 1.0f;
  const float inv = 1.0f / am1;
  float* row = sc + (size_t)b * L_SZ;

  float lmax = -INFINITY;
  for (int l = tid; l < L_SZ; l += 256) {
    const float x = row[l] * am1;
    sXs[l] = x;
    lmax = fmaxf(lmax, x);
  }
  #pragma unroll
  for (int o = 16; o > 0; o >>= 1) lmax = fmaxf(lmax, __shfl_xor(lmax, o, 32));
  if ((tid & 31) == 0) red[tid >> 5] = lmax;
  __syncthreads();
  if (tid < 32) {
    float x = (tid < 8) ? red[tid] : -INFINITY;
    #pragma unroll
    for (int o = 4; o > 0; o >>= 1) x = fmaxf(x, __shfl_xor(x, o, 32));
    if (tid == 0) bc = x;
  }
  __syncthreads();
  const float mx = bc;

  float tau_lo = mx - 1.0f;                       // gp(1, alpha) = 1
  const float tau_hi = mx - exp2f(-12.0f * am1);  // (1/4096)^(alpha-1)
  float dm = tau_hi - tau_lo;

  // p(tau) element: clip(Xs - tau, 0)^(1/(alpha-1))
  auto pw = [inv](float x) -> float {
    return x > 0.0f ? exp2f(inv * log2f(x)) : 0.0f;
  };

  float s = 0.0f;
  for (int l = tid; l < L_SZ; l += 256) s += pw(sXs[l] - tau_lo);
  const float f_lo = blk_sum(s, red, &bc, tid) - 1.0f;

  float tau_m = tau_lo;
  for (int it = 0; it < NITER; ++it) {
    dm *= 0.5f;
    tau_m = tau_lo + dm;
    float s2 = 0.0f;
    for (int l = tid; l < L_SZ; l += 256) s2 += pw(sXs[l] - tau_m);
    const float f_m = blk_sum(s2, red, &bc, tid) - 1.0f;
    if (f_m * f_lo >= 0.0f) tau_lo = tau_m;       // uniform across block
  }

  float s3 = 0.0f;
  for (int l = tid; l < L_SZ; l += 256) {
    const float p = pw(sXs[l] - tau_m);
    sXs[l] = p;
    s3 += p;
  }
  const float S = blk_sum(s3, red, &bc, tid);
  const float rinv = 1.0f / S;
  for (int l = tid; l < L_SZ; l += 256) row[l] = sXs[l] * rinv;
}

// ---------------------------------------------------------------------------
// Kernel 3: part[b][chunk][d] = sum over 256 rows of a[b,l]*R[b,l,d].
// Entmax weights are sparse: zero-weight rows skip the R load entirely
// (uniform branch -> whole-wave skip), cutting most of the 2nd R stream.
// Deterministic LDS tree reduction (no float atomics).
// ---------------------------------------------------------------------------
__global__ __launch_bounds__(256) void k_wsum(
    const float* __restrict__ R, const float* __restrict__ a,
    float* __restrict__ part)
{
  __shared__ float sAcc[8][128];
  const int b = blockIdx.y, chunk = blockIdx.x;
  const int tid = threadIdx.x, wave = tid >> 5, lane = tid & 31;
  const int lbase = chunk * 256 + wave * 32;
  const float* arow = a + (size_t)b * L_SZ + lbase;
  const float* Rb   = R + ((size_t)b * L_SZ + lbase) * D_SZ;

  float4 acc = make_float4(0.f, 0.f, 0.f, 0.f);
  for (int r = 0; r < 32; ++r) {
    const float w = arow[r];       // uniform across the wave
    if (w != 0.0f) {
      const float4 v = *(const float4*)(Rb + (size_t)r * D_SZ + lane * 4);
      acc.x += w * v.x; acc.y += w * v.y; acc.z += w * v.z; acc.w += w * v.w;
    }
  }
  sAcc[wave][lane * 4 + 0] = acc.x;
  sAcc[wave][lane * 4 + 1] = acc.y;
  sAcc[wave][lane * 4 + 2] = acc.z;
  sAcc[wave][lane * 4 + 3] = acc.w;
  __syncthreads();
  if (tid < 128) {
    float sum = 0.0f;
    #pragma unroll
    for (int w = 0; w < 8; ++w) sum += sAcc[w][tid];
    part[((size_t)b * NCHUNK + chunk) * D_SZ + tid] = sum;
  }
}

// ---------------------------------------------------------------------------
// Kernel 4: sum partials, selu, L2 normalize.
// ---------------------------------------------------------------------------
__global__ __launch_bounds__(128) void k_final(
    const float* __restrict__ part, float* __restrict__ out)
{
  __shared__ float red[4];
  __shared__ float bc;
  const int b = blockIdx.x, tid = threadIdx.x;
  float s = 0.0f;
  #pragma unroll
  for (int c = 0; c < NCHUNK; ++c) s += part[((size_t)b * NCHUNK + c) * D_SZ + tid];

  const float kScale = 1.0507009873554805f;
  const float kAlpha = 1.6732632423543772f;
  const float y = s > 0.0f ? kScale * s : kScale * kAlpha * expm1f(s);

  float q = warp_sum(y * y);
  if ((tid & 31) == 0) red[tid >> 5] = q;
  __syncthreads();
  if (tid < 32) {
    float x = (tid < 4) ? red[tid] : 0.0f;
    x = warp_sum(x);
    if (tid == 0) bc = x;
  }
  __syncthreads();
  out[b * D_SZ + tid] = y / sqrtf(bc);
}

// ---------------------------------------------------------------------------
extern "C" void kernel_launch(void* const* d_in, const int* in_sizes, int n_in,
                              void* d_out, int out_size, void* d_ws, size_t ws_size,
                              hipStream_t stream) {
  (void)in_sizes; (void)n_in; (void)out_size; (void)ws_size;
  const float* R      = (const float*)d_in[0];
  const float* alpha  = (const float*)d_in[1];
  const float* target = (const float*)d_in[2];
  const float* Wr1    = (const float*)d_in[3];
  const float* Wr2    = (const float*)d_in[4];
  const float* Wr3    = (const float*)d_in[5];
  const float* bias_r = (const float*)d_in[6];
  const float* Wf_w   = (const float*)d_in[7];
  const float* Wf_b   = (const float*)d_in[8];

  float* ws     = (float*)d_ws;
  float* c_buf  = ws;                                  // B*D      = 16384
  float* sc_buf = ws + B_SZ * D_SZ;                    // B*L      = 524288 (scores -> a in place)
  float* part   = sc_buf + (size_t)B_SZ * L_SZ;        // B*16*D   = 262144
  float* out    = (float*)d_out;

  k_prep  <<<dim3(B_SZ),             dim3(128), 0, stream>>>(target, Wr2, bias_r, Wf_w, Wf_b, c_buf);
  k_scores<<<dim3(L_SZ/128, B_SZ),   dim3(256), 0, stream>>>(R, Wr1, Wr3, c_buf, sc_buf);
  k_entmax<<<dim3(B_SZ),             dim3(256), 0, stream>>>(alpha, sc_buf);
  k_wsum  <<<dim3(NCHUNK, B_SZ),     dim3(256), 0, stream>>>(R, sc_buf, part);
  k_final <<<dim3(B_SZ),             dim3(128), 0, stream>>>(part, out);
}
